// MultiHeadSelfAttention_41875931136470
// MI455X (gfx1250) — compile-verified, matrix operand order checked by reading, not run on hardware
//
#include <hip/hip_runtime.h>

// ---------------------------------------------------------------------------
// MultiHeadSelfAttention (GPT-NeoX style, full rotary, causal) for gfx1250.
// bf16 WMMA (V_WMMA_F32_16X16X32_BF16) for all matrix math, f32 accumulate.
// Projection GEMMs: 64x128 macro-tiles, double-buffered TDM staging into LDS.
// ---------------------------------------------------------------------------

typedef __bf16 bf16_t;
typedef __attribute__((ext_vector_type(16))) __bf16 v16bf;
typedef __attribute__((ext_vector_type(8)))  float  v8f;
typedef __attribute__((ext_vector_type(4)))  unsigned int v4u;
typedef __attribute__((ext_vector_type(8)))  int v8i;
typedef __attribute__((ext_vector_type(4)))  int v4i;

#define B_  2
#define S_  2048
#define D_  1024
#define H_  16
#define HD_ 64

// ---- WMMA wrapper ----------------------------------------------------------
__device__ __forceinline__ v8f wmma_bf16(v16bf a, v16bf b, v8f c) {
  return __builtin_amdgcn_wmma_f32_16x16x32_bf16(false, a, false, b,
                                                 (short)0, c, false, false);
}

// ---- Fragment loaders (16-bit A/B layouts per CDNA5 ISA 7.12.2) -----------
// k = kBase + g*16 + half*8 + j  (g=0..1, j=0..7); lane&15 = row/col.
__device__ __forceinline__ v16bf load_frag_rowmajor(const bf16_t* base, int ld,
                                                    int rowBase, int kBase,
                                                    int lane) {
  int m  = lane & 15;
  int hl = lane >> 4;
  const bf16_t* p = base + (size_t)(rowBase + m) * ld + kBase + hl * 8;
  v16bf a;
#pragma unroll
  for (int g = 0; g < 2; ++g)
#pragma unroll
    for (int j = 0; j < 8; ++j)
      a[g * 8 + j] = p[g * 16 + j];
  return a;
}

// V fragment for P*V: B[k][n] = V[jBase+k][dcol], V row-major [S][HD].
__device__ __forceinline__ v16bf load_vfrag(const bf16_t* V, int jBase,
                                            int dcol, int lane) {
  int hl = lane >> 4;
  v16bf a;
#pragma unroll
  for (int g = 0; g < 2; ++g)
#pragma unroll
    for (int j = 0; j < 8; ++j) {
      int kl = g * 16 + hl * 8 + j;
      a[g * 8 + j] = V[(size_t)(jBase + kl) * HD_ + dcol];
    }
  return a;
}

// ---- TDM: async DMA of a (tileRows x 32) bf16 tile (row-major, ld=K) -------
// D# per CDNA5 ISA 8.3/8.4.
__device__ __forceinline__ void tdm_load_tile(unsigned ldsOff,
                                              const bf16_t* gsrc, int K,
                                              int nRows, int rowBase, int kBase,
                                              int tileRows) {
  unsigned long long addr =
      (unsigned long long)(const void*)(gsrc + (size_t)rowBase * K + kBase);
  v4u g0;
  g0[0] = 1u;                                         // count=1, user mode
  g0[1] = ldsOff;                                     // lds_addr (bytes)
  g0[2] = (unsigned)(addr & 0xffffffffu);             // global_addr[31:0]
  g0[3] = (unsigned)((addr >> 32) & 0x01ffffffu) |    // global_addr[56:32]
          (2u << 30);                                 // type=2 ("image")
  v8i g1;
  g1[0] = (int)(1u << 16);                            // wg_mask=0, data_size=2B
  g1[1] = (int)(((unsigned)K & 0xffffu) << 16);       // tensor_dim0[15:0]
  g1[2] = (int)((((unsigned)K >> 16) & 0xffffu) |     // tensor_dim0[31:16]
                (((unsigned)nRows & 0xffffu) << 16)); // tensor_dim1[15:0]
  g1[3] = (int)((((unsigned)nRows >> 16) & 0xffffu) | // tensor_dim1[31:16]
                (32u << 16));                         // tile_dim0 = 32 elems
  g1[4] = tileRows;                                   // tile_dim1
  g1[5] = K;                                          // dim0_stride[31:0]
  g1[6] = 0;                                          // dim0_stride[47:32]
  g1[7] = 0;
  v4i gz = {0, 0, 0, 0};
#if __clang_major__ >= 23
  v8i gz8 = {0, 0, 0, 0, 0, 0, 0, 0};
  __builtin_amdgcn_tensor_load_to_lds(g0, g1, gz, gz, gz8, 0);
#else
  __builtin_amdgcn_tensor_load_to_lds(g0, g1, gz, gz, 0);
#endif
}

// ---- 16-lane row reductions (rows of C layout span 16 lanes per half) -----
__device__ __forceinline__ float rowmax16(float v) {
  v = fmaxf(v, __shfl_xor(v, 1, 32));
  v = fmaxf(v, __shfl_xor(v, 2, 32));
  v = fmaxf(v, __shfl_xor(v, 4, 32));
  v = fmaxf(v, __shfl_xor(v, 8, 32));
  return v;
}
__device__ __forceinline__ float rowsum16(float v) {
  v += __shfl_xor(v, 1, 32);
  v += __shfl_xor(v, 2, 32);
  v += __shfl_xor(v, 4, 32);
  v += __shfl_xor(v, 8, 32);
  return v;
}

// ---------------------------------------------------------------------------
// fp32 -> bf16 conversion
// ---------------------------------------------------------------------------
__global__ void cvt_f32_bf16_kernel(const float* __restrict__ src,
                                    bf16_t* __restrict__ dst, int n) {
  for (int i = blockIdx.x * blockDim.x + threadIdx.x; i < n;
       i += gridDim.x * blockDim.x)
    dst[i] = (bf16_t)src[i];
}

// ---------------------------------------------------------------------------
// Macro-tiled GEMM core: 4 waves, 64(M) x 128(N) tile, K staged through LDS
// by the TDM with double buffering (next tile's DMA overlaps current WMMAs;
// s_wait_tensorcnt<=2 completes only the older pair of tensor ops).
// A is [Mtot][K] row-major, W is [Ntot][K] row-major (WMMA B operand = W^T,
// same K-contiguous fragment pattern). Wave w owns rows [w*16, +16); 8 column
// sub-tiles => 8 WMMAs per K-step per wave.
// ---------------------------------------------------------------------------
#define NT_ 8
#define ATILE_ (64 * 32)
#define BTILE_ (NT_ * 16 * 32)

__device__ __forceinline__ void wave_gemm_core_tdm(
    const bf16_t* __restrict__ A, const bf16_t* __restrict__ W, int K,
    int Mtot, int Ntot, int rowBase, int colBase, bf16_t* ldsA, bf16_t* ldsB,
    v8f acc[NT_]) {
  int lane = threadIdx.x & 31;
  int wave = threadIdx.x >> 5;

  v8f zero = {};
#pragma unroll
  for (int i = 0; i < NT_; ++i) acc[i] = zero;

  unsigned offA0 = (unsigned)(unsigned long long)(void*)ldsA;
  unsigned offB0 = (unsigned)(unsigned long long)(void*)ldsB;

  int nIter = K / 32;
  if (wave == 0) {
    tdm_load_tile(offA0, A, K, Mtot, rowBase, 0, 64);
    tdm_load_tile(offB0, W, K, Ntot, colBase, 0, NT_ * 16);
  }

  for (int i = 0; i < nIter; ++i) {
    int cur = i & 1;
    int nxt = cur ^ 1;
    if (wave == 0) {
      if (i + 1 < nIter) {
        tdm_load_tile(offA0 + (unsigned)(nxt * ATILE_ * 2), A, K, Mtot, rowBase,
                      (i + 1) * 32, 64);
        tdm_load_tile(offB0 + (unsigned)(nxt * BTILE_ * 2), W, K, Ntot, colBase,
                      (i + 1) * 32, NT_ * 16);
        __builtin_amdgcn_s_wait_tensorcnt((short)2);  // older pair done
      } else {
        __builtin_amdgcn_s_wait_tensorcnt((short)0);
      }
    }
    __syncthreads();  // current tiles visible to all waves

    const bf16_t* curA = ldsA + cur * ATILE_;
    const bf16_t* curB = ldsB + cur * BTILE_;

    v16bf a = load_frag_rowmajor(curA, 32, wave * 16, 0, lane);
    v16bf bw[NT_];
#pragma unroll
    for (int nt = 0; nt < NT_; ++nt)
      bw[nt] = load_frag_rowmajor(curB, 32, nt * 16, 0, lane);
#pragma unroll
    for (int nt = 0; nt < NT_; ++nt) acc[nt] = wmma_bf16(a, bw[nt], acc[nt]);

    __syncthreads();  // reads done before this buffer is overwritten
  }
}

// ---------------------------------------------------------------------------
// QKV projection: qkv[row, col] = sum_k x[row,k] * W_qkv[col,k]
// Epilogue scatters into [B,H,S,64] q/k/v (bf16); each 16-wide column
// sub-tile lies in a single (three, h) slice.
// ---------------------------------------------------------------------------
__global__ __launch_bounds__(128) void gemm_qkv_kernel(
    const bf16_t* __restrict__ X, const bf16_t* __restrict__ Wq,
    bf16_t* __restrict__ qA, bf16_t* __restrict__ kA, bf16_t* __restrict__ vA) {
  __shared__ bf16_t ldsA[2 * ATILE_];
  __shared__ bf16_t ldsB[2 * BTILE_];

  int lane = threadIdx.x & 31;
  int wave = threadIdx.x >> 5;
  int l16 = lane & 15, hl = lane >> 4;
  int rowBase = blockIdx.x * 64;
  int colBase = blockIdx.y * 128;

  v8f acc[NT_];
  wave_gemm_core_tdm(X, Wq, D_, B_ * S_, 3 * D_, rowBase, colBase, ldsA, ldsB,
                     acc);

#pragma unroll
  for (int nt = 0; nt < NT_; ++nt) {
    int colT = colBase + nt * 16;       // 16-aligned -> one (three,h) slice
    int three = colT >> 10;
    int h     = (colT >> 6) & (H_ - 1);
    int hd    = (colT & 63) + l16;
    bf16_t* dst = (three == 0) ? qA : ((three == 1) ? kA : vA);
#pragma unroll
    for (int r = 0; r < 8; ++r) {
      int row = rowBase + wave * 16 + r + 8 * hl;  // C layout M = r+8*half
      int b = row >> 11;                           // row = b*S + s
      int s = row & (S_ - 1);
      dst[(((size_t)b * H_ + h) * S_ + s) * HD_ + hd] = (bf16_t)acc[nt][r];
    }
  }
}

// ---------------------------------------------------------------------------
// Output projection: out[row, col] = sum_k attO[row,k] * W_out[col,k] (fp32)
// ---------------------------------------------------------------------------
__global__ __launch_bounds__(128) void gemm_out_kernel(
    const bf16_t* __restrict__ A, const bf16_t* __restrict__ Wo,
    float* __restrict__ out) {
  __shared__ bf16_t ldsA[2 * ATILE_];
  __shared__ bf16_t ldsB[2 * BTILE_];

  int lane = threadIdx.x & 31;
  int wave = threadIdx.x >> 5;
  int l16 = lane & 15, hl = lane >> 4;
  int rowBase = blockIdx.x * 64;
  int colBase = blockIdx.y * 128;

  v8f acc[NT_];
  wave_gemm_core_tdm(A, Wo, D_, B_ * S_, D_, rowBase, colBase, ldsA, ldsB, acc);

#pragma unroll
  for (int r = 0; r < 8; ++r) {
    int row = rowBase + wave * 16 + r + 8 * hl;
#pragma unroll
    for (int nt = 0; nt < NT_; ++nt)
      out[(size_t)row * D_ + colBase + nt * 16 + l16] = acc[nt][r];
  }
}

// ---------------------------------------------------------------------------
// Rotary embedding (full head dim), in place on bf16 [B,H,S,64] q and k.
// ---------------------------------------------------------------------------
__global__ void rotary_kernel(bf16_t* __restrict__ qA, bf16_t* __restrict__ kA) {
  int idx = blockIdx.x * blockDim.x + threadIdx.x;  // 2^21 threads total
  int i = idx & 31;
  int s = (idx >> 5) & (S_ - 1);
  int h = (idx >> 16) & (H_ - 1);
  int b = idx >> 20;

  float freq = __powf(10000.0f, -((float)(2 * i)) / 64.0f);
  float ang = (float)s * freq;
  float c = __cosf(ang), sn = __sinf(ang);

  size_t base = (((size_t)b * H_ + h) * S_ + s) * HD_;
  {
    float x1 = (float)qA[base + i], x2 = (float)qA[base + i + 32];
    qA[base + i]      = (bf16_t)(x1 * c - x2 * sn);
    qA[base + i + 32] = (bf16_t)(x2 * c + x1 * sn);
  }
  {
    float x1 = (float)kA[base + i], x2 = (float)kA[base + i + 32];
    kA[base + i]      = (bf16_t)(x1 * c - x2 * sn);
    kA[base + i + 32] = (bf16_t)(x2 * c + x1 * sn);
  }
}

// ---------------------------------------------------------------------------
// Flash attention: one wave per (b, h, 16-query tile); 32 keys per step.
// ---------------------------------------------------------------------------
__global__ __launch_bounds__(32) void flash_attn_kernel(
    const bf16_t* __restrict__ qA, const bf16_t* __restrict__ kA,
    const bf16_t* __restrict__ vA, const int* __restrict__ amask,
    bf16_t* __restrict__ attO) {
  __shared__ bf16_t ldsP[16 * 32];

  int lane = threadIdx.x;
  int l16 = lane & 15, hl = lane >> 4;

  int idx = blockIdx.x;  // B*H*(S/16) = 4096
  int qt = idx & 127;
  int h  = (idx >> 7) & (H_ - 1);
  int b  = idx >> 11;
  int qBase = qt * 16;

  const bf16_t* Q  = qA + (((size_t)b * H_ + h) * S_) * HD_;
  const bf16_t* Kp = kA + (((size_t)b * H_ + h) * S_) * HD_;
  const bf16_t* Vp = vA + (((size_t)b * H_ + h) * S_) * HD_;

  // Q fragments (A layout), pre-scaled by 1/sqrt(HD) = 0.125 (exact in bf16).
  v16bf qf[2];
#pragma unroll
  for (int t = 0; t < 2; ++t) {
    v16bf raw = load_frag_rowmajor(Q, HD_, qBase, t * 32, lane);
#pragma unroll
    for (int e = 0; e < 16; ++e) raw[e] = (bf16_t)((float)raw[e] * 0.125f);
    qf[t] = raw;
  }

  float mr[8], lr[8];
  v8f zero = {};
  v8f o[4];
#pragma unroll
  for (int i = 0; i < 4; ++i) o[i] = zero;
#pragma unroll
  for (int r = 0; r < 8; ++r) { mr[r] = -1e30f; lr[r] = 0.0f; }

  int jtMax = (qBase + 15) >> 5;  // causal: last 32-key block needed
  for (int jt = 0; jt <= jtMax; ++jt) {
    int jBase = jt * 32;

    // ---- scores: S[16q x 32k] (load all K fragments, then WMMA burst) ----
    v16bf k00 = load_frag_rowmajor(Kp, HD_, jBase,      0,  lane);
    v16bf k01 = load_frag_rowmajor(Kp, HD_, jBase,      32, lane);
    v16bf k10 = load_frag_rowmajor(Kp, HD_, jBase + 16, 0,  lane);
    v16bf k11 = load_frag_rowmajor(Kp, HD_, jBase + 16, 32, lane);
    v8f s0 = zero, s1 = zero;
    s0 = wmma_bf16(qf[0], k00, s0);
    s0 = wmma_bf16(qf[1], k01, s0);
    s1 = wmma_bf16(qf[0], k10, s1);
    s1 = wmma_bf16(qf[1], k11, s1);

    // ---- mask + online softmax (per C-layout row) ----
    int key0 = jBase + l16;
    int key1 = jBase + 16 + l16;
    bool pad0 = (amask[b * S_ + key0] == 0);
    bool pad1 = (amask[b * S_ + key1] == 0);

    float p0a[8], p1a[8];
#pragma unroll
    for (int r = 0; r < 8; ++r) {
      int qg = qBase + r + 8 * hl;
      float v0 = s0[r], v1 = s1[r];
      if (key0 > qg || pad0) v0 = -1e30f;
      if (key1 > qg || pad1) v1 = -1e30f;

      float rm = rowmax16(fmaxf(v0, v1));
      float mnew = fmaxf(mr[r], rm);
      float alpha = __expf(mr[r] - mnew);
      float p0 = (v0 <= -1e29f) ? 0.0f : __expf(v0 - mnew);
      float p1 = (v1 <= -1e29f) ? 0.0f : __expf(v1 - mnew);
      float rs = rowsum16(p0 + p1);

      lr[r] = lr[r] * alpha + rs;
      mr[r] = mnew;
#pragma unroll
      for (int nt = 0; nt < 4; ++nt) o[nt][r] *= alpha;
      p0a[r] = p0;
      p1a[r] = p1;
    }

    // ---- P: C layout -> A layout through LDS ----
    __syncthreads();
#pragma unroll
    for (int r = 0; r < 8; ++r) {
      int row = r + 8 * hl;
      ldsP[row * 32 + l16]      = (bf16_t)p0a[r];
      ldsP[row * 32 + 16 + l16] = (bf16_t)p1a[r];
    }
    __syncthreads();
    v16bf pf = load_frag_rowmajor(ldsP, 32, 0, 0, lane);

    // ---- O += P * V (load all V fragments, then WMMA burst) ----
    v16bf vf[4];
#pragma unroll
    for (int nt = 0; nt < 4; ++nt)
      vf[nt] = load_vfrag(Vp, jBase, nt * 16 + l16, lane);
#pragma unroll
    for (int nt = 0; nt < 4; ++nt) o[nt] = wmma_bf16(pf, vf[nt], o[nt]);
  }

  // ---- finalize: divide by l, write to [B,S,H*64] (already transposed) ----
#pragma unroll
  for (int r = 0; r < 8; ++r) {
    float inv = 1.0f / lr[r];
    int srow = qBase + r + 8 * hl;
    size_t base = ((size_t)b * S_ + srow) * D_ + h * HD_;
#pragma unroll
    for (int nt = 0; nt < 4; ++nt)
      attO[base + nt * 16 + l16] = (bf16_t)(o[nt][r] * inv);
  }
}

// ---------------------------------------------------------------------------
// Launch
// ---------------------------------------------------------------------------
extern "C" void kernel_launch(void* const* d_in, const int* in_sizes, int n_in,
                              void* d_out, int out_size, void* d_ws,
                              size_t ws_size, hipStream_t stream) {
  const float* x     = (const float*)d_in[0];
  const int*   amask = (const int*)d_in[1];
  const float* wqkv  = (const float*)d_in[2];
  const float* wout  = (const float*)d_in[3];
  float* out = (float*)d_out;

  char* ws = (char*)d_ws;
  size_t off = 0;
  auto alloc_bf16 = [&](size_t elems) {
    bf16_t* p = (bf16_t*)(ws + off);
    off += ((elems * sizeof(bf16_t) + 255) & ~(size_t)255);
    return p;
  };
  bf16_t* xb    = alloc_bf16((size_t)B_ * S_ * D_);
  bf16_t* wqkvb = alloc_bf16((size_t)3 * D_ * D_);
  bf16_t* woutb = alloc_bf16((size_t)D_ * D_);
  bf16_t* qA    = alloc_bf16((size_t)B_ * H_ * S_ * HD_);
  bf16_t* kA    = alloc_bf16((size_t)B_ * H_ * S_ * HD_);
  bf16_t* vA    = alloc_bf16((size_t)B_ * H_ * S_ * HD_);
  bf16_t* attO  = alloc_bf16((size_t)B_ * S_ * D_);
  (void)ws_size; (void)n_in; (void)in_sizes; (void)out_size;

  // 1) fp32 -> bf16 operand conversion
  cvt_f32_bf16_kernel<<<1024, 256, 0, stream>>>(x, xb, B_ * S_ * D_);
  cvt_f32_bf16_kernel<<<1024, 256, 0, stream>>>(wqkv, wqkvb, 3 * D_ * D_);
  cvt_f32_bf16_kernel<<<1024, 256, 0, stream>>>(wout, woutb, D_ * D_);

  // 2) QKV projection (double-buffered TDM, scatters into q/k/v [B,H,S,64])
  gemm_qkv_kernel<<<dim3((B_ * S_) / 64, (3 * D_) / 128), 128, 0, stream>>>(
      xb, wqkvb, qA, kA, vA);

  // 3) rotary on q, k
  rotary_kernel<<<(B_ * H_ * S_ * 32) / 256, 256, 0, stream>>>(qA, kA);

  // 4) flash attention -> attO [B,S,D] bf16
  flash_attn_kernel<<<B_ * H_ * (S_ / 16), 32, 0, stream>>>(qA, kA, vA, amask,
                                                            attO);

  // 5) output projection (double-buffered TDM) -> fp32 d_out
  gemm_out_kernel<<<dim3((B_ * S_) / 64, D_ / 128), 128, 0, stream>>>(attO,
                                                                      woutb,
                                                                      out);
}
